// Attention_42339787604185
// MI455X (gfx1250) — matrix operand order repaired and checked by reading, not compiled, not run on hardware
//
#include <hip/hip_runtime.h>
#include <type_traits>

#define DEVINL static __device__ __forceinline__

typedef _Float16 v16h __attribute__((ext_vector_type(16)));
typedef _Float16 v8h  __attribute__((ext_vector_type(8)));
typedef _Float16 v4h  __attribute__((ext_vector_type(4)));
typedef float    v8f  __attribute__((ext_vector_type(8)));

constexpr int kB = 2, kT = 2048, kC = 1024, kH = 16, kHD = 64;
constexpr int kK = kC;                                    // GEMM reduction depth
constexpr size_t kHeadElems = (size_t)kB * kH * kT * kHD; // 4M halves = 8 MB

DEVINL v16h make16(v8h lo, v8h hi) {
  return __builtin_shufflevector(lo, hi, 0,1,2,3,4,5,6,7,8,9,10,11,12,13,14,15);
}
DEVINL v8f wmma32(v16h a, v16h b, v8f c) {
  // (neg_a, A, neg_b, B, c_mod, C, reuse_a, reuse_b)
  return __builtin_amdgcn_wmma_f32_16x16x32_f16(false, a, false, b, (short)0, c, false, false);
}

// ---------------------------------------------------------------------------
// Tiled WMMA GEMM: [M=4096, K=1024] x [K=1024, N] -> f32, A optionally f16.
// Block = 256 threads (8 waves), BM=32, BN=64, BK=32, one 16x16 tile / wave.
// SCATTER: q/k as f16 [B,H,T,HD]; v as f16 [B,H,HD,T] (transposed for P*V).
// ---------------------------------------------------------------------------
template<int N, bool AHALF, bool SCATTER>
__global__ __launch_bounds__(256)
void gemm_kernel(const void* __restrict__ Ap, const float* __restrict__ Bw,
                 float* __restrict__ outF,
                 _Float16* __restrict__ qf, _Float16* __restrict__ kf,
                 _Float16* __restrict__ vf)
{
  __shared__ __align__(16) _Float16 lA[32 * 32];      // [m][k], stride 32
  __shared__ __align__(16) _Float16 lB[64 * 40];      // [n][k], stride 40 (padded)

  const int tid  = threadIdx.x;
  const int lane = tid & 31;
  const int wave = tid >> 5;
  const int ln = lane & 15, lh = lane >> 4;
  const int wm = (wave & 1) * 16;     // wave's M offset within block tile
  const int wn = (wave >> 1) * 16;    // wave's N offset within block tile
  const int gm0 = blockIdx.y * 32;
  const int gn0 = blockIdx.x * 64;

  v8f acc = {};

  for (int k0 = 0; k0 < kK; k0 += 32) {
    __syncthreads();
    { // stage A: 32 rows x 32 k, 4 halves per thread, coalesced
      const int row = tid >> 3;
      const int kq  = (tid & 7) * 4;
      if constexpr (AHALF) {
        const _Float16* A = (const _Float16*)Ap;
        *(v4h*)&lA[row * 32 + kq] =
            *(const v4h*)&A[(size_t)(gm0 + row) * kK + k0 + kq];
      } else {
        const float* A = (const float*)Ap;
        const float4 f = *(const float4*)&A[(size_t)(gm0 + row) * kK + k0 + kq];
        v4h h4 = { (_Float16)f.x, (_Float16)f.y, (_Float16)f.z, (_Float16)f.w };
        *(v4h*)&lA[row * 32 + kq] = h4;
      }
    }
    { // stage B transposed: 32 k x 64 n, coalesced global reads
      const int n  = tid & 63;
      const int kb = (tid >> 6) * 8;
      #pragma unroll
      for (int kk = 0; kk < 8; ++kk)
        lB[n * 40 + kb + kk] =
            (_Float16)Bw[(size_t)(k0 + kb + kk) * N + gn0 + n];
    }
    __syncthreads();

    // A fragment: lane holds row wm+ln; k = lh*8 + i (i<8), 16+lh*8+(i-8) (i>=8)
    const _Float16* ar = &lA[(wm + ln) * 32 + lh * 8];
    const v16h a = make16(*(const v8h*)ar, *(const v8h*)(ar + 16));
    // B fragment: lane holds col wn+ln; k = lh*8 + i (i<8), 16+lh*8+(i-8) (i>=8)
    const _Float16* br = &lB[(wn + ln) * 40 + lh * 8];
    const v16h b = make16(*(const v8h*)br, *(const v8h*)(br + 16));
    acc = wmma32(a, b, acc);
  }

  // D layout: element r of lane -> (m = r + 8*lh, n = ln) within the tile
  #pragma unroll
  for (int r = 0; r < 8; ++r) {
    const int gr = gm0 + wm + r + 8 * lh;   // global row (b*T + t)
    const int gc = gn0 + wn + ln;           // global col
    if constexpr (SCATTER) {
      const int bb = gr >> 11, tt = gr & (kT - 1);
      const int sec = gc >> 10, cc = gc & (kC - 1);
      const int hh = cc >> 6, dd = cc & (kHD - 1);
      const _Float16 hv = (_Float16)acc[r];
      if (sec == 2)   // V transposed: [B,H,HD,T]
        vf[(((size_t)bb * kH + hh) * kHD + dd) * kT + tt] = hv;
      else {          // Q/K: [B,H,T,HD]
        _Float16* dst = (sec == 0) ? qf : kf;
        dst[(((size_t)bb * kH + hh) * kT + tt) * kHD + dd] = hv;
      }
    } else {
      outF[(size_t)gr * N + gc] = acc[r];
    }
  }
}

// ---------------------------------------------------------------------------
// RoPE (roll variant): out[d] = x[d]*cos(t*w[d]) + x[(d-1) mod 64]*sin(t*w[d])
// One thread owns one whole (b,h,t) row of q or k -> in-place, race-free.
// ---------------------------------------------------------------------------
__global__ __launch_bounds__(256)
void rope_kernel(_Float16* __restrict__ qf, _Float16* __restrict__ kf)
{
  const int idx = blockIdx.x * blockDim.x + threadIdx.x;  // 0 .. 2*B*H*T-1
  const int row = idx & (kB * kH * kT - 1);               // 65536 rows per matrix
  const int mtx = idx >> 16;
  _Float16* p = (mtx ? kf : qf) + (size_t)row * kHD;
  const int t = row & (kT - 1);

  float xv[kHD];
  #pragma unroll
  for (int d = 0; d < kHD; ++d) xv[d] = (float)p[d];

  const float ln10000 = 9.21034037197618f;
  #pragma unroll
  for (int d = 0; d < kHD; ++d) {
    const float inv = __expf(-(float)(2 * (d & 31)) * (1.0f / 64.0f) * ln10000);
    const float ang = (float)t * inv;
    float s, c;
    __sincosf(ang, &s, &c);
    p[d] = (_Float16)(xv[d] * c + xv[(d + kHD - 1) & (kHD - 1)] * s);
  }
}

// ---------------------------------------------------------------------------
// Flash attention: 1 wave per 16-query tile per (b,h). Key blocks of 32.
// S = Q K^T via 4 WMMAs, online softmax (log2 domain, bare v_exp),
// O += P V via 4 WMMAs.  Causal mask applied only on diagonal tail blocks.
// ---------------------------------------------------------------------------
__global__ __launch_bounds__(32)
void attn_kernel(const _Float16* __restrict__ Q, const _Float16* __restrict__ K,
                 const _Float16* __restrict__ Vt, _Float16* __restrict__ Y)
{
  __shared__ __align__(16) _Float16 lP[16 * 32];   // P tile, [query m][key kk]

  const int bh = blockIdx.y;                 // b*H + h
  const int q0 = blockIdx.x * 16;
  const int lane = threadIdx.x;
  const int ln = lane & 15, lh = lane >> 4;

  const _Float16* Qb = Q  + (size_t)bh * kT * kHD;
  const _Float16* Kb = K  + (size_t)bh * kT * kHD;
  const _Float16* Vb = Vt + (size_t)bh * kHD * kT;   // [d][t]

  // Q A-fragments (row = query q0+ln), chunks d=[0,32) and d=[32,64)
  const _Float16* qr = &Qb[(size_t)(q0 + ln) * kHD + lh * 8];
  const v16h qa0 = make16(*(const v8h*)qr,        *(const v8h*)(qr + 16));
  const v16h qa1 = make16(*(const v8h*)(qr + 32), *(const v8h*)(qr + 48));

  float rmax[8], rsum[8];
  v8f o0 = {}, o1 = {}, o2 = {}, o3 = {};
  #pragma unroll
  for (int r = 0; r < 8; ++r) { rmax[r] = -1e30f; rsum[r] = 0.0f; }

  // logits kept in log2 domain: s * (1/sqrt(64)) * log2(e)
  const float kSc = 0.125f * 1.44269504088896f;

  auto process_block = [&](int jb, auto maskedTag) {
    constexpr bool MASKED = decltype(maskedTag)::value;

    // ---- S tiles: s0 = keys [jb, jb+16), s1 = keys [jb+16, jb+32) ----
    v8f s0 = {}, s1 = {};
    {
      const _Float16* kp = &Kb[(size_t)(jb + ln) * kHD + lh * 8];
      s0 = wmma32(qa0, make16(*(const v8h*)kp,        *(const v8h*)(kp + 16)),  s0);
      s0 = wmma32(qa1, make16(*(const v8h*)(kp + 32), *(const v8h*)(kp + 48)), s0);
      const _Float16* kp2 = &Kb[(size_t)(jb + 16 + ln) * kHD + lh * 8];
      s1 = wmma32(qa0, make16(*(const v8h*)kp2,        *(const v8h*)(kp2 + 16)),  s1);
      s1 = wmma32(qa1, make16(*(const v8h*)(kp2 + 32), *(const v8h*)(kp2 + 48)), s1);
    }

    // ---- online softmax (rows live in 16-lane halves) ----
    #pragma unroll
    for (int r = 0; r < 8; ++r) {
      const int m = r + 8 * lh;
      float a0 = s0[r] * kSc;
      float a1 = s1[r] * kSc;
      if constexpr (MASKED) {
        const int iq = q0 + m;
        a0 = (jb + ln      <= iq) ? a0 : -1e30f;
        a1 = (jb + 16 + ln <= iq) ? a1 : -1e30f;
      }
      float bm = fmaxf(a0, a1);
      bm = fmaxf(bm, __shfl_xor(bm, 1, 32));
      bm = fmaxf(bm, __shfl_xor(bm, 2, 32));
      bm = fmaxf(bm, __shfl_xor(bm, 4, 32));
      bm = fmaxf(bm, __shfl_xor(bm, 8, 32));
      const float mn = fmaxf(rmax[r], bm);
      const float al = __builtin_amdgcn_exp2f(rmax[r] - mn);
      rmax[r] = mn;
      const float p0 = __builtin_amdgcn_exp2f(a0 - mn);
      const float p1 = __builtin_amdgcn_exp2f(a1 - mn);
      float ps = p0 + p1;
      ps += __shfl_xor(ps, 1, 32);
      ps += __shfl_xor(ps, 2, 32);
      ps += __shfl_xor(ps, 4, 32);
      ps += __shfl_xor(ps, 8, 32);
      rsum[r] = rsum[r] * al + ps;
      o0[r] *= al; o1[r] *= al; o2[r] *= al; o3[r] *= al;
      lP[m * 32 + ln]      = (_Float16)p0;   // D-layout -> LDS (A-layout source)
      lP[m * 32 + 16 + ln] = (_Float16)p1;
    }

    // ---- P A-fragment from LDS (intra-wave DS ops are in-order) ----
    const _Float16* pr = &lP[ln * 32 + lh * 8];
    const v16h pa = make16(*(const v8h*)pr, *(const v8h*)(pr + 16));

    // ---- O += P x V : four 16-wide HD tiles, vector loads from Vt ----
    #pragma unroll
    for (int t4 = 0; t4 < 4; ++t4) {
      // b[i] = V[key = jb + 8*lh + i (i<8), jb + 16 + 8*lh + (i-8) (i>=8)][d = t4*16 + ln] = Vt[d][key]
      const _Float16* vp = &Vb[(size_t)(t4 * 16 + ln) * kT + jb + lh * 8];
      const v16h vbv = make16(*(const v8h*)vp, *(const v8h*)(vp + 16));
      v8f& o = (t4 == 0) ? o0 : (t4 == 1) ? o1 : (t4 == 2) ? o2 : o3;
      o = wmma32(pa, vbv, o);
    }
  };

  // Full (unmasked) blocks: jb + 31 <= q0  ->  jb < jfull
  const int jfull = (q0 >= 32) ? (((q0 - 31) >> 5) << 5) + 32 : 0;
  const int jend  = q0 + 16;                 // exclusive causal key bound
  for (int jb = 0; jb < jfull; jb += 32)
    process_block(jb, std::integral_constant<bool, false>{});
  for (int jb = jfull; jb < jend; jb += 32)
    process_block(jb, std::integral_constant<bool, true>{});

  // ---- normalize and store y in [B,T,C] (f16 for the proj GEMM) ----
  const int b = bh >> 4, h = bh & 15;
  #pragma unroll
  for (int r = 0; r < 8; ++r) {
    const int m = r + 8 * lh;
    const float inv = 1.0f / rsum[r];
    _Float16* yrow = &Y[(size_t)(b * kT + q0 + m) * kC + h * kHD + ln];
    yrow[0]  = (_Float16)(o0[r] * inv);
    yrow[16] = (_Float16)(o1[r] * inv);
    yrow[32] = (_Float16)(o2[r] * inv);
    yrow[48] = (_Float16)(o3[r] * inv);
  }
}

// ---------------------------------------------------------------------------
extern "C" void kernel_launch(void* const* d_in, const int* in_sizes, int n_in,
                              void* d_out, int out_size, void* d_ws, size_t ws_size,
                              hipStream_t stream)
{
  (void)in_sizes; (void)n_in; (void)out_size; (void)ws_size;
  const float* x  = (const float*)d_in[0];   // [B,T,C]
  const float* Wa = (const float*)d_in[1];   // [C,3C]
  const float* Wp = (const float*)d_in[2];   // [C,C]
  // d_in[3] = causal mask: implemented analytically, unused.
  float* out = (float*)d_out;                // [B,T,C] f32

  _Float16* qf = (_Float16*)d_ws;            // [B,H,T,HD], 8 MB
  _Float16* kf = qf + kHeadElems;            // [B,H,T,HD], 8 MB
  _Float16* vf = kf + kHeadElems;            // [B,H,HD,T], 8 MB (transposed)
  _Float16* yf = vf + kHeadElems;            // [B,T,C],    8 MB

  // 1) qkv = x @ W_attn, scattered as f16 into head-major q/k and transposed v
  gemm_kernel<3 * kC, false, true>
      <<<dim3((3 * kC) / 64, (kB * kT) / 32), 256, 0, stream>>>(
          (const void*)x, Wa, nullptr, qf, kf, vf);

  // 2) RoPE in place on q and k
  rope_kernel<<<dim3((2 * kB * kH * kT) / 256), 256, 0, stream>>>(qf, kf);

  // 3) causal flash attention -> y [B,T,C] f16
  attn_kernel<<<dim3(kT / 16, kB * kH), 32, 0, stream>>>(qf, kf, vf, yf);

  // 4) out = y @ W_proj (f32 output)
  gemm_kernel<kC, true, false>
      <<<dim3(kC / 64, (kB * kT) / 32), 256, 0, stream>>>(
          (const void*)yf, Wp, out, nullptr, nullptr, nullptr);
}
